// ATTR_TRANSFORMER_6811818132216
// MI455X (gfx1250) — matrix operand order repaired and checked by reading, not compiled, not run on hardware
//
#include <hip/hip_runtime.h>
#include <hip/hip_bf16.h>
#include <math.h>

typedef __attribute__((ext_vector_type(16))) _Float16 v16h;
typedef __attribute__((ext_vector_type(8)))  _Float16 v8h;
typedef __attribute__((ext_vector_type(8)))  float    v8f;
typedef __attribute__((ext_vector_type(4)))  float    v4f;

#define B_  4096
#define S_  128
#define D_  128
#define H_  128
#define U_  256
#define AXP 132   // padded row stride (floats) for transposed attn_x in LDS

// ---------------- ws layout ----------------
// [0,       32768)  : W_attr as f16   (H*D halfs)
// [32768,  163840)  : attn_user f32   (U*H floats)
// [163840, 180224)  : b_to_u  int32   (B ints)

__device__ __forceinline__ float fast_tanh(float v) {
#if __has_builtin(__builtin_amdgcn_tanhf)
    return __builtin_amdgcn_tanhf(v);           // V_TANH_F32
#else
    // tanh(v) = 1 - 2/(exp2(2v*log2e)+1)
    float e = __builtin_amdgcn_exp2f(v * 2.8853900817779268f);
    return 1.0f - 2.0f / (e + 1.0f);
#endif
}

__device__ __forceinline__ float fast_exp(float v) {
#if __has_builtin(__builtin_amdgcn_exp2f)
    return __builtin_amdgcn_exp2f(v * 1.4426950408889634f);  // V_EXP_F32
#else
    return expf(v);
#endif
}

__global__ void k_convert_wattr(const float* __restrict__ W_attr,
                                _Float16* __restrict__ wh) {
    int i = blockIdx.x * blockDim.x + threadIdx.x;
    if (i < H_ * D_) wh[i] = (_Float16)W_attr[i];
}

// attn_user[u][h] = b_user[h] + sum_d user[u][d] * W_user[h][d]
__global__ void k_attn_user(const float* __restrict__ user,
                            const float* __restrict__ W_user,
                            const float* __restrict__ b_user,
                            float* __restrict__ au) {
    __shared__ float su[D_];
    int u = blockIdx.x;
    int t = threadIdx.x;           // t = h, blockDim = 128
    su[t] = user[u * D_ + t];
    __syncthreads();
    float acc = b_user[t];
    const float* wr = W_user + t * D_;
    for (int d = 0; d < D_; ++d) acc += su[d] * wr[d];
    au[u * H_ + t] = acc;
}

// b_to_u[b] = u such that b falls in u's repeat range (prefix sums of user_size)
__global__ void k_b2u(const int* __restrict__ user_size, int* __restrict__ b2u) {
    __shared__ int ssz[U_];
    int t = threadIdx.x;           // blockDim = U_
    ssz[t] = user_size[t];
    __syncthreads();
    int start = 0;
    for (int v = 0; v < t; ++v) start += ssz[v];
    int n = ssz[t];
    for (int i = 0; i < n; ++i) {
        int bpos = start + i;
        if (bpos < B_) b2u[bpos] = t;
    }
}

// One block per batch element b. 128 threads = 4 waves.
__global__ void __launch_bounds__(128)
__attribute__((amdgpu_waves_per_eu(4)))     // cap at 256 VGPRs: no VGPR-MSB churn
k_main(const float* __restrict__ x, const float* __restrict__ x_mask,
       const _Float16* __restrict__ wh, const float* __restrict__ au,
       const int* __restrict__ b2u, const float* __restrict__ b_attr,
       float* __restrict__ out_wx, float* __restrict__ out_w) {
    // ax is TRANSPOSED: ax[h * AXP + s] = attn_x[s][h]
    __shared__ __align__(16) float ax[H_ * AXP];
    __shared__ __align__(16) float ru[H_];
    __shared__ __align__(16) float red[S_];
    __shared__ float bb[H_];
    __shared__ float wred[4];

    const int b    = blockIdx.x;
    const int t    = threadIdx.x;        // 0..127
    const int lane = t & 31;
    const int w    = t >> 5;             // wave id 0..3
    const int lr   = lane & 15;          // row/col within 16-tile
    const int lh   = lane >> 4;          // half-wave select

    const float* xb = x + (size_t)b * S_ * D_;

    const int uidx = b2u[b];
    ru[t] = au[uidx * H_ + t];
    bb[t] = b_attr[t];
    __syncthreads();

    // ---- GEMM: attn_x = tanh(x[b] @ W_attr^T + b_attr), f16 WMMA ----
    // Wave w owns M-tiles {2w, 2w+1}; each B fragment feeds both (halves B traffic).
    const int mt0 = 2 * w, mt1 = 2 * w + 1;
    v8f acc0[8] = {};
    v8f acc1[8] = {};
    #pragma unroll
    for (int ks = 0; ks < 4; ++ks) {
        // A fragments (16x32 f16): lane holds K = lh*8..lh*8+7 and +16
        const v4f* pa0 = (const v4f*)(xb + (mt0 * 16 + lr) * D_ + ks * 32 + lh * 8);
        const v4f* pa1 = (const v4f*)(xb + (mt1 * 16 + lr) * D_ + ks * 32 + lh * 8);
        const v4f a00 = pa0[0], a01 = pa0[1], a02 = pa0[4], a03 = pa0[5];
        const v4f a10 = pa1[0], a11 = pa1[1], a12 = pa1[4], a13 = pa1[5];
        v16h af0, af1;
        #pragma unroll
        for (int i = 0; i < 4; ++i) {
            af0[i]      = (_Float16)a00[i];
            af0[4 + i]  = (_Float16)a01[i];
            af0[8 + i]  = (_Float16)a02[i];
            af0[12 + i] = (_Float16)a03[i];
            af1[i]      = (_Float16)a10[i];
            af1[4 + i]  = (_Float16)a11[i];
            af1[8 + i]  = (_Float16)a12[i];
            af1[12 + i] = (_Float16)a13[i];
        }
        #pragma unroll
        for (int nt = 0; nt < 8; ++nt) {
            // B fragment (32x16 f16): lane n holds K = lh*8..lh*8+7 and +16
            const v8h* pb = (const v8h*)(wh + (nt * 16 + lr) * D_ + ks * 32 + lh * 8);
            const v8h b0 = pb[0], b1 = pb[2];
            v16h bf;
            #pragma unroll
            for (int i = 0; i < 8; ++i) { bf[i] = b0[i]; bf[8 + i] = b1[i]; }
            acc0[nt] = __builtin_amdgcn_wmma_f32_16x16x32_f16(
                false, af0, false, bf, (short)0, acc0[nt], false, false);
            acc1[nt] = __builtin_amdgcn_wmma_f32_16x16x32_f16(
                false, af1, false, bf, (short)0, acc1[nt], false, false);
        }
    }
    // Epilogue: C layout VGPR r -> M = r + 8*lh, N = lr.
    // Transposed ax => 8 consecutive s per tile => two b128 stores.
    #pragma unroll
    for (int nt = 0; nt < 8; ++nt) {
        const int col = nt * 16 + lr;          // h index
        const float bias = bb[col];
        v4f t0, t1;
        #pragma unroll
        for (int r = 0; r < 4; ++r) {
            t0[r] = fast_tanh(acc0[nt][r] + bias);
            t1[r] = fast_tanh(acc0[nt][4 + r] + bias);
        }
        v4f* dst0 = (v4f*)&ax[col * AXP + mt0 * 16 + 8 * lh];
        dst0[0] = t0; dst0[1] = t1;
        #pragma unroll
        for (int r = 0; r < 4; ++r) {
            t0[r] = fast_tanh(acc1[nt][r] + bias);
            t1[r] = fast_tanh(acc1[nt][4 + r] + bias);
        }
        v4f* dst1 = (v4f*)&ax[col * AXP + mt1 * 16 + 8 * lh];
        dst1[0] = t0; dst1[1] = t1;
    }
    __syncthreads();

    // ---- score[s] = mask[s] * dot_h(attn_x[s], ru) ; thread t -> s = t ----
    const float msk = x_mask[(size_t)b * S_ + t];
    float dot = 0.0f;
    #pragma unroll 4
    for (int h = 0; h < H_; ++h) dot += ax[h * AXP + t] * ru[h];
    const float sco = dot * msk;

    // ---- softmax over s: wave32 shuffle trees + 4-way cross-wave combine ----
    float vmax = sco;
    #pragma unroll
    for (int off = 16; off > 0; off >>= 1)
        vmax = fmaxf(vmax, __shfl_xor(vmax, off, 32));
    if (lane == 0) wred[w] = vmax;
    __syncthreads();
    const float m = fmaxf(fmaxf(wred[0], wred[1]), fmaxf(wred[2], wred[3]));
    const float e = fast_exp(sco - m);
    __syncthreads();                       // all reads of wred done before reuse
    float vsum = e;
    #pragma unroll
    for (int off = 16; off > 0; off >>= 1)
        vsum += __shfl_xor(vsum, off, 32);
    if (lane == 0) wred[w] = vsum;
    __syncthreads();
    const float den = (wred[0] + wred[1]) + (wred[2] + wred[3]);
    const float aw = e / den;
    out_w[(size_t)b * S_ + t] = aw;        // attn_weight output
    red[t] = aw * msk;                     // attn_weight * mask for step 6
    __syncthreads();

    // ---- attn_weighted_x[b][h] = sum_s attn_x[s][h] * (aw*mask)[s] ----
    // Transposed ax => contiguous row read, vectorized b128.
    const v4f* rowp = (const v4f*)&ax[t * AXP];
    const v4f* wp   = (const v4f*)red;
    v4f accv = {};
    #pragma unroll 8
    for (int s4 = 0; s4 < S_ / 4; ++s4) {
        const v4f a = rowp[s4];
        const v4f wv = wp[s4];
        accv += a * wv;
    }
    out_wx[(size_t)b * H_ + t] = (accv[0] + accv[1]) + (accv[2] + accv[3]);
}

extern "C" void kernel_launch(void* const* d_in, const int* in_sizes, int n_in,
                              void* d_out, int out_size, void* d_ws, size_t ws_size,
                              hipStream_t stream) {
    const float* x      = (const float*)d_in[0];
    const float* x_mask = (const float*)d_in[1];
    const float* user   = (const float*)d_in[2];
    const int*   usz    = (const int*)d_in[3];
    const float* W_attr = (const float*)d_in[4];
    const float* b_attr = (const float*)d_in[5];
    const float* W_user = (const float*)d_in[6];
    const float* b_user = (const float*)d_in[7];

    _Float16* wh  = (_Float16*)d_ws;
    float*    au  = (float*)((char*)d_ws + 32768);
    int*      b2u = (int*)((char*)d_ws + 163840);

    float* out_wx = (float*)d_out;                       // [B,H]
    float* out_w  = (float*)d_out + (size_t)B_ * H_;     // [B,S]

    k_convert_wattr<<<(H_ * D_ + 255) / 256, 256, 0, stream>>>(W_attr, wh);
    k_attn_user<<<U_, 128, 0, stream>>>(user, W_user, b_user, au);
    k_b2u<<<1, U_, 0, stream>>>(usz, b2u);
    k_main<<<B_, 128, 0, stream>>>(x, x_mask, wh, au, b2u, b_attr, out_wx, out_w);
}
